// AnyGPTAttentionOnly_1348619730973
// MI455X (gfx1250) — matrix operand order repaired and checked by reading, not compiled, not run on hardware
//
#include <hip/hip_runtime.h>
#include <hip/hip_bf16.h>

// ---------------------------------------------------------------------------
// GPT-Neo local attention block, bf16 WMMA pipeline for gfx1250 (MI455X).
// B=4, S=2048, H=1024, NH=16, HD=64, WIN=256.
// ---------------------------------------------------------------------------

#define BB   4
#define SS   2048
#define HH   1024
#define NHH  16
#define HDD  64
#define WINN 256
#define MM   (BB * SS)          // 8192 rows

typedef unsigned short ushort_t;
typedef __attribute__((ext_vector_type(16))) __bf16 v16bf;
typedef __attribute__((ext_vector_type(8)))  float  v8f;

union FragU {
    v16bf v;
    struct { uint4 lo; uint4 hi; } q;
};

static __device__ __forceinline__ ushort_t f2bf(float f) {
    unsigned int u = __float_as_uint(f);
    unsigned int r = (u + 0x7FFFu + ((u >> 16) & 1u)) >> 16;   // RNE
    return (ushort_t)r;
}

static __device__ __forceinline__ v8f wmma_bf16(v16bf a, v16bf b, v8f c) {
    // D(16x16 f32) = A(16x32 bf16) * B(32x16 bf16) + C
    return __builtin_amdgcn_wmma_f32_16x16x32_bf16(
        false, a, false, b, (short)0, c, false, false);
}

// ---------------------------------------------------------------------------
// Stage 1: f32 -> bf16 cast (weights)
// ---------------------------------------------------------------------------
__global__ void cast_bf16_kernel(const float* __restrict__ src,
                                 ushort_t* __restrict__ dst, int n) {
    int i = blockIdx.x * blockDim.x + threadIdx.x;
    int stride = gridDim.x * blockDim.x;
    for (; i < n; i += stride) dst[i] = f2bf(src[i]);
}

// ---------------------------------------------------------------------------
// Stage 2: LayerNorm (residual, f32) + cast raw x -> bf16. One row per block.
// ---------------------------------------------------------------------------
__global__ __launch_bounds__(256)
void ln_cast_kernel(const float* __restrict__ x,
                    const float* __restrict__ lnw,
                    const float* __restrict__ lnb,
                    float* __restrict__ resid,
                    ushort_t* __restrict__ xb) {
    __shared__ float s1[256], s2[256];
    const int row = blockIdx.x;
    const int tid = threadIdx.x;
    const float* xr = x + (size_t)row * HH;

    float4 v = ((const float4*)xr)[tid];                    // 4 elems / thread
    float sum = v.x + v.y + v.z + v.w;
    float sq  = v.x * v.x + v.y * v.y + v.z * v.z + v.w * v.w;
    s1[tid] = sum; s2[tid] = sq;
    __syncthreads();
    for (int off = 128; off > 0; off >>= 1) {
        if (tid < off) { s1[tid] += s1[tid + off]; s2[tid] += s2[tid + off]; }
        __syncthreads();
    }
    const float mu  = s1[0] * (1.0f / HH);
    const float var = s2[0] * (1.0f / HH) - mu * mu;
    const float rs  = rsqrtf(var + 1e-7f);

    const float vv[4] = { v.x, v.y, v.z, v.w };
#pragma unroll
    for (int j = 0; j < 4; ++j) {
        int col = tid * 4 + j;
        size_t idx = (size_t)row * HH + col;
        resid[idx] = (vv[j] - mu) * rs * lnw[col] + lnb[col];
        xb[idx]    = f2bf(vv[j]);                           // raw x for QKV
    }
}

// ---------------------------------------------------------------------------
// Stage 3/5: WMMA GEMM  out[m,n] = sum_k A[m,k] * W[n,k]
//   MODE 0: bf16 out in [B,NH,S,HD] layout        (Q, K)
//   MODE 1: bf16 out in [B,NH,HD,S] layout        (V transposed)
//   MODE 2: f32 out = acc + bias[n] + resid[m,n]  (final projection)
// One wave computes a 64(M) x 64(N) tile: 4 A-frags x 4 B-frags = 16 WMMA
// per 32-wide k-step -> each fragment reused 4x (bytes/FLOP cut 4x vs 16x64).
// blockDim = 128 (4 waves).
// ---------------------------------------------------------------------------
template <int MODE>
__global__ __launch_bounds__(128)
void gemm_kernel(const ushort_t* __restrict__ A,   // [M, 1024] bf16
                 const ushort_t* __restrict__ W,   // [1024, 1024] bf16 (N-major)
                 ushort_t* __restrict__ outb,
                 const float* __restrict__ bias,
                 const float* __restrict__ resid,
                 float* __restrict__ outf) {
    const int lane = threadIdx.x & 31;
    const int wid  = threadIdx.x >> 5;
    const int flat = blockIdx.x * 4 + wid;          // 2048 wave tiles
    const int mt64 = flat >> 4;                     // 128 M-tiles of 64
    const int nt   = flat & 15;                     // 16  N-tiles of 64
    const int m0 = mt64 << 6, n0 = nt << 6;
    const int half = lane >> 4, nl = lane & 15;

    v8f acc[4][4];
#pragma unroll
    for (int mt = 0; mt < 4; ++mt)
#pragma unroll
        for (int t = 0; t < 4; ++t)
#pragma unroll
            for (int r = 0; r < 8; ++r) acc[mt][t][r] = 0.0f;

    const ushort_t* arow0 = A + (size_t)(m0 + nl) * HH;

    for (int kk = 0; kk < HH; kk += 32) {
        FragU a[4];                                 // A 16x32 frags: row=nl
#pragma unroll
        for (int mt = 0; mt < 4; ++mt) {
            const ushort_t* ar = arow0 + (size_t)(mt * 16) * HH + kk;
            a[mt].q.lo = *(const uint4*)(ar + half * 8);
            a[mt].q.hi = *(const uint4*)(ar + 16 + half * 8);
            __builtin_prefetch(ar + 32, 0, 0);      // next k-slice of A
        }
#pragma unroll
        for (int t = 0; t < 4; ++t) {
            const ushort_t* wrow =
                W + (size_t)(n0 + t * 16 + nl) * HH + kk + half * 8;
            FragU b;                                // B 32x16: col=nl, k=half*8+i / 16+half*8+i
            b.q.lo = ((const uint4*)wrow)[0];
            b.q.hi = ((const uint4*)wrow)[2];
#pragma unroll
            for (int mt = 0; mt < 4; ++mt)
                acc[mt][t] = wmma_bf16(a[mt].v, b.v, acc[mt][t]);
        }
    }

#pragma unroll
    for (int mt = 0; mt < 4; ++mt) {
#pragma unroll
        for (int t = 0; t < 4; ++t) {
#pragma unroll
            for (int r = 0; r < 8; ++r) {
                const int m = m0 + mt * 16 + r + 8 * half;  // C: M in vgpr/half
                const int n = n0 + t * 16 + nl;             //    N in lane
                if (MODE == 2) {
                    size_t idx = (size_t)m * HH + n;
                    outf[idx] = acc[mt][t][r] + bias[n] + resid[idx];
                } else {
                    const int b_ = m >> 11, s_ = m & (SS - 1);
                    const int h_ = n >> 6,  d_ = n & (HDD - 1);
                    if (MODE == 0)
                        outb[((size_t)(b_ * NHH + h_) * SS + s_) * HDD + d_] =
                            f2bf(acc[mt][t][r]);
                    else
                        outb[((size_t)(b_ * NHH + h_) * HDD + d_) * SS + s_] =
                            f2bf(acc[mt][t][r]);
                }
            }
        }
    }
}

// ---------------------------------------------------------------------------
// Stage 4: banded local attention (flash-style, online softmax).
// One wave per 16-query tile; key tiles of 32 (all 32-aligned, no OOB).
// Q,K: [B,NH,S,64] bf16.  Vt: [B,NH,64,S] bf16.  ctx out: [B,S,H] bf16.
// ---------------------------------------------------------------------------
__global__ __launch_bounds__(128)
void attn_kernel(const ushort_t* __restrict__ qb,
                 const ushort_t* __restrict__ kb,
                 const ushort_t* __restrict__ vtb,
                 ushort_t* __restrict__ ctxb) {
    __shared__ __align__(16) ushort_t p_lds[4][16][32];   // per-wave P tile

    const int lane = threadIdx.x & 31;
    const int wid  = threadIdx.x >> 5;
    const int flat = blockIdx.x * 4 + wid;      // 8192 waves
    const int bh = flat >> 7;                   // 64 (b,h) pairs
    const int qt = flat & 127;
    const int q0 = qt << 4;
    const int half = lane >> 4, nl = lane & 15;

    const ushort_t* qp  = qb  + ((size_t)bh * SS + q0) * HDD;
    const ushort_t* kp  = kb  + (size_t)bh * SS * HDD;
    const ushort_t* vtp = vtb + (size_t)bh * HDD * SS;

    // Q A-fragments for d=[0,32) and [32,64): row = nl per lane
    FragU aq0, aq1;
    {
        const ushort_t* qr = qp + nl * HDD;
        aq0.q.lo = *(const uint4*)(qr + half * 8);
        aq0.q.hi = *(const uint4*)(qr + 16 + half * 8);
        aq1.q.lo = *(const uint4*)(qr + 32 + half * 8);
        aq1.q.hi = *(const uint4*)(qr + 48 + half * 8);
    }

    float row_max[8], row_sum[8];
    v8f o[4];
#pragma unroll
    for (int r = 0; r < 8; ++r) { row_max[r] = -3.0e38f; row_sum[r] = 0.0f; }
#pragma unroll
    for (int t = 0; t < 4; ++t)
#pragma unroll
        for (int r = 0; r < 8; ++r) o[t][r] = 0.0f;

    const int jlo = (q0 > WINN - 1) ? ((q0 - (WINN - 1)) & ~31) : 0;
    const int jhi = q0 + 15;

    for (int jt = jlo; jt <= jhi; jt += 32) {
        // ---- S = Q K^T for two 16-key halves --------------------------------
        v8f sc[2];
#pragma unroll
        for (int kh = 0; kh < 2; ++kh) {
            const int key = jt + kh * 16 + nl;          // B col = lane
            const ushort_t* krow = kp + (size_t)key * HDD;
            FragU bk0, bk1;                             // k rows = half*8+i / 16+half*8+i
            bk0.q.lo = ((const uint4*)(krow + half * 8))[0];
            bk0.q.hi = ((const uint4*)(krow + half * 8))[2];
            bk1.q.lo = ((const uint4*)(krow + 32 + half * 8))[0];
            bk1.q.hi = ((const uint4*)(krow + 32 + half * 8))[2];
            v8f z;
#pragma unroll
            for (int r = 0; r < 8; ++r) z[r] = 0.0f;
            z = wmma_bf16(aq0.v, bk0.v, z);
            sc[kh] = wmma_bf16(aq1.v, bk1.v, z);
        }

        // ---- mask + online softmax update ----------------------------------
#pragma unroll
        for (int r = 0; r < 8; ++r) {
            const int qi = q0 + r + 8 * half;
            const int k0 = jt + nl, k1 = jt + 16 + nl;
            const bool v0 = (k0 <= qi) && (qi - k0 < WINN);
            const bool v1 = (k1 <= qi) && (qi - k1 < WINN);
            float s0 = v0 ? sc[0][r] : -3.0e38f;
            float s1 = v1 ? sc[1][r] : -3.0e38f;

            float mx = fmaxf(s0, s1);
            mx = fmaxf(mx, __shfl_xor(mx, 1, 32));
            mx = fmaxf(mx, __shfl_xor(mx, 2, 32));
            mx = fmaxf(mx, __shfl_xor(mx, 4, 32));
            mx = fmaxf(mx, __shfl_xor(mx, 8, 32));     // 16-lane row reduce

            const float mnew  = fmaxf(row_max[r], mx);
            const float alpha = __expf(row_max[r] - mnew);
            const float p0 = v0 ? __expf(s0 - mnew) : 0.0f;
            const float p1 = v1 ? __expf(s1 - mnew) : 0.0f;

            p_lds[wid][r + 8 * half][nl]      = f2bf(p0);
            p_lds[wid][r + 8 * half][16 + nl] = f2bf(p1);

            float ps = p0 + p1;
            ps += __shfl_xor(ps, 1, 32);
            ps += __shfl_xor(ps, 2, 32);
            ps += __shfl_xor(ps, 4, 32);
            ps += __shfl_xor(ps, 8, 32);

            row_sum[r] = row_sum[r] * alpha + ps;
            row_max[r] = mnew;
#pragma unroll
            for (int t = 0; t < 4; ++t) o[t][r] *= alpha;
        }

        asm volatile("s_wait_dscnt 0" ::: "memory");
        __builtin_amdgcn_wave_barrier();

        // ---- P (A-frag 16x32) from LDS, O += P * V -------------------------
        FragU ap;
        {
            const ushort_t* prow = &p_lds[wid][nl][0];
            ap.q.lo = *(const uint4*)(prow + half * 8);
            ap.q.hi = *(const uint4*)(prow + 16 + half * 8);
        }
#pragma unroll
        for (int t = 0; t < 4; ++t) {
            const ushort_t* vrow =
                vtp + (size_t)(t * 16 + nl) * SS + jt + half * 8;
            FragU bv;
            bv.q.lo = ((const uint4*)vrow)[0];
            bv.q.hi = ((const uint4*)vrow)[2];
            o[t] = wmma_bf16(ap.v, bv.v, o[t]);
        }
        __builtin_amdgcn_wave_barrier();
    }

    // ---- normalize + store ctx as bf16 [B,S,H] -----------------------------
    const int b_ = bh >> 4, h_ = bh & 15;
#pragma unroll
    for (int r = 0; r < 8; ++r) {
        const float inv = 1.0f / row_sum[r];
        const int s_ = q0 + r + 8 * half;
#pragma unroll
        for (int t = 0; t < 4; ++t) {
            ctxb[((size_t)(b_ * SS + s_)) * HH + h_ * HDD + t * 16 + nl] =
                f2bf(o[t][r] * inv);
        }
    }
}

// ---------------------------------------------------------------------------
// Host launcher
// ---------------------------------------------------------------------------
extern "C" void kernel_launch(void* const* d_in, const int* in_sizes, int n_in,
                              void* d_out, int out_size, void* d_ws, size_t ws_size,
                              hipStream_t stream) {
    const float* x    = (const float*)d_in[0];
    const float* wq   = (const float*)d_in[1];
    const float* wk   = (const float*)d_in[2];
    const float* wv   = (const float*)d_in[3];
    const float* wo   = (const float*)d_in[4];
    const float* bo   = (const float*)d_in[5];
    const float* ln_w = (const float*)d_in[6];
    const float* ln_b = (const float*)d_in[7];
    float* out = (float*)d_out;

    char* ws = (char*)d_ws;
    const size_t MB = 1024 * 1024;
    ushort_t* xb    = (ushort_t*)(ws + 0);          // 16 MB
    ushort_t* wqb   = (ushort_t*)(ws + 16 * MB);    //  2 MB
    ushort_t* wkb   = (ushort_t*)(ws + 18 * MB);
    ushort_t* wvb   = (ushort_t*)(ws + 20 * MB);
    ushort_t* wob   = (ushort_t*)(ws + 22 * MB);
    ushort_t* qbuf  = (ushort_t*)(ws + 24 * MB);    // 16 MB  [B,NH,S,64]
    ushort_t* kbuf  = (ushort_t*)(ws + 40 * MB);    // 16 MB  [B,NH,S,64]
    ushort_t* vtbuf = (ushort_t*)(ws + 56 * MB);    // 16 MB  [B,NH,64,S]
    ushort_t* ctxb  = (ushort_t*)(ws + 72 * MB);    // 16 MB  [B,S,H]
    float*    resid = (float*)   (ws + 88 * MB);    // 32 MB

    // 1) weights -> bf16
    cast_bf16_kernel<<<512, 256, 0, stream>>>(wq, wqb, HH * HH);
    cast_bf16_kernel<<<512, 256, 0, stream>>>(wk, wkb, HH * HH);
    cast_bf16_kernel<<<512, 256, 0, stream>>>(wv, wvb, HH * HH);
    cast_bf16_kernel<<<512, 256, 0, stream>>>(wo, wob, HH * HH);

    // 2) LayerNorm residual + x -> bf16
    ln_cast_kernel<<<MM, 256, 0, stream>>>(x, ln_w, ln_b, resid, xb);

    // 3) QKV projections (head-split layouts), 512 blocks x 4 waves
    gemm_kernel<0><<<512, 128, 0, stream>>>(xb, wqb, qbuf,  nullptr, nullptr, nullptr);
    gemm_kernel<0><<<512, 128, 0, stream>>>(xb, wkb, kbuf,  nullptr, nullptr, nullptr);
    gemm_kernel<1><<<512, 128, 0, stream>>>(xb, wvb, vtbuf, nullptr, nullptr, nullptr);

    // 4) banded local attention
    attn_kernel<<<2048, 128, 0, stream>>>(qbuf, kbuf, vtbuf, ctxb);

    // 5) output projection + bias + residual
    gemm_kernel<2><<<512, 128, 0, stream>>>(ctxb, wob, nullptr, bo, resid, out);
}
